// SelfAttention_43688407335034
// MI455X (gfx1250) — compile-verified, matrix operand order checked by reading, not run on hardware
//
#include <hip/hip_runtime.h>

// ---------------- problem constants ----------------
#define BB 4
#define SS 2048
#define DD 1024
#define HH 16
#define DHH 64          // DD / HH
#define E3 (3 * DD)     // 3072

typedef __attribute__((ext_vector_type(16))) __bf16       v16bf;
typedef __attribute__((ext_vector_type(8)))  float        v8f;
typedef __attribute__((ext_vector_type(8)))  unsigned int v8u;
typedef __attribute__((ext_vector_type(4)))  unsigned int v4u;
typedef __attribute__((ext_vector_type(8)))  int          i32x8;
typedef __attribute__((ext_vector_type(4)))  int          i32x4;

#define HD static __device__ __forceinline__

// round-to-nearest-even fp32 -> bf16 (bit pattern)
HD unsigned short f2bf(float f) {
  unsigned u = __builtin_bit_cast(unsigned, f);
  unsigned r = u + 0x7FFFu + ((u >> 16) & 1u);
  return (unsigned short)(r >> 16);
}

HD v8f zero8() {
  v8f z;
#pragma unroll
  for (int i = 0; i < 8; ++i) z[i] = 0.0f;
  return z;
}

HD v8f wmma_bf16(v16bf a, v16bf b, v8f c) {
  return __builtin_amdgcn_wmma_f32_16x16x32_bf16(false, a, false, b, (short)0, c,
                                                 false, false);
}

// ---- DPP 16-lane row reductions (pure VALU, dual-issues with the rest) -----
HD float row_max16(float x) {
  float y;
  y = __builtin_bit_cast(float, __builtin_amdgcn_update_dpp(0, __builtin_bit_cast(int, x), 0xB1, 0xF, 0xF, true));
  x = fmaxf(x, y);
  y = __builtin_bit_cast(float, __builtin_amdgcn_update_dpp(0, __builtin_bit_cast(int, x), 0x4E, 0xF, 0xF, true));
  x = fmaxf(x, y);
  y = __builtin_bit_cast(float, __builtin_amdgcn_update_dpp(0, __builtin_bit_cast(int, x), 0x141, 0xF, 0xF, true));
  x = fmaxf(x, y);
  y = __builtin_bit_cast(float, __builtin_amdgcn_update_dpp(0, __builtin_bit_cast(int, x), 0x140, 0xF, 0xF, true));
  return fmaxf(x, y);
}
HD float row_sum16(float x) {
  float y;
  y = __builtin_bit_cast(float, __builtin_amdgcn_update_dpp(0, __builtin_bit_cast(int, x), 0xB1, 0xF, 0xF, true));
  x += y;
  y = __builtin_bit_cast(float, __builtin_amdgcn_update_dpp(0, __builtin_bit_cast(int, x), 0x4E, 0xF, 0xF, true));
  x += y;
  y = __builtin_bit_cast(float, __builtin_amdgcn_update_dpp(0, __builtin_bit_cast(int, x), 0x141, 0xF, 0xF, true));
  x += y;
  y = __builtin_bit_cast(float, __builtin_amdgcn_update_dpp(0, __builtin_bit_cast(int, x), 0x140, 0xF, 0xF, true));
  return x + y;
}

// ---- fragment loads (ISA 16-bit A layout: two contiguous 16B runs/lane) ----
HD v16bf load_frag_rm(const unsigned short* p, int ld, int r0, int k0) {
  int lane = threadIdx.x & 31;
  int r = lane & 15, kh = lane >> 4;
  const unsigned short* base = p + (size_t)(r0 + r) * ld + (k0 + 8 * kh);
  v4u lo = *(const v4u*)(base);
  v4u hi = *(const v4u*)(base + 16);
  v8u u;
  u[0] = lo[0]; u[1] = lo[1]; u[2] = lo[2]; u[3] = lo[3];
  u[4] = hi[0]; u[5] = hi[1]; u[6] = hi[2]; u[7] = hi[3];
  return __builtin_bit_cast(v16bf, u);
}

// B fragment from K-major storage: element (k,n) at p[(k0+k)*ld + n0+n]
HD v16bf load_bfrag_kn(const unsigned short* p, int ld, int k0, int n0) {
  int lane = threadIdx.x & 31;
  int n = lane & 15, kh = lane >> 4;
  v8u u;
#pragma unroll
  for (int v = 0; v < 8; ++v) {
    int k = ((v < 4) ? (2 * v) : (2 * v + 8)) + 8 * kh;
    unsigned e0 = p[(size_t)(k0 + k) * ld + n0 + n];
    unsigned e1 = p[(size_t)(k0 + k + 1) * ld + n0 + n];
    u[v] = e0 | (e1 << 16);
  }
  return __builtin_bit_cast(v16bf, u);
}

// ---- Tensor Data Mover: 2-D tile (tile_d0 contiguous elems x tile_d1 rows) -
HD unsigned lds_off(const void* p) {
  return (unsigned)(unsigned long long)(size_t)p;  // flat LDS addr low 32 bits
}

HD void tdm_load_2d(unsigned lds_addr, const void* gptr, unsigned tile_d0,
                    unsigned tile_d1, unsigned long long stride0_elems) {
  unsigned long long ga = (unsigned long long)(size_t)gptr;
  v4u g0;
  g0[0] = 1u;                                              // count=1, user desc
  g0[1] = lds_addr;                                        // lds_addr [63:32]
  g0[2] = (unsigned)ga;                                    // global_addr lo
  g0[3] = (unsigned)((ga >> 32) & 0x01FFFFFFu) | (2u << 30);  // addr hi | type=2
  i32x8 g1;
  g1[0] = (int)(1u << 16);                                 // data_size=1 -> 2B
  g1[1] = (int)((tile_d0 & 0xFFFFu) << 16);                // tensor_dim0 lo16
  g1[2] = (int)(((tile_d0 >> 16) & 0xFFFFu) | ((tile_d1 & 0xFFFFu) << 16));
  g1[3] = (int)(((tile_d1 >> 16) & 0xFFFFu) | ((tile_d0 & 0xFFFFu) << 16));  // tile_dim0
  g1[4] = (int)(tile_d1 & 0xFFFFu);                        // tile_dim1 (dim2=0)
  g1[5] = (int)(unsigned)(stride0_elems & 0xFFFFFFFFu);    // dim0 stride lo
  g1[6] = (int)(unsigned)((stride0_elems >> 32) & 0xFFFFu);
  g1[7] = 0;
  i32x4 z4 = {0, 0, 0, 0};
#if __clang_major__ >= 23
  i32x8 z8 = {0, 0, 0, 0, 0, 0, 0, 0};
  __builtin_amdgcn_tensor_load_to_lds(g0, g1, z4, z4, z8, 0);
#else
  __builtin_amdgcn_tensor_load_to_lds(g0, g1, z4, z4, 0);
#endif
}

// ---------------- fp32 -> bf16 conversion ----------------
__global__ void cvt_f32_bf16(const float* __restrict__ src,
                             unsigned short* __restrict__ dst, int n) {
  int i = blockIdx.x * blockDim.x + threadIdx.x;
  int stride = gridDim.x * blockDim.x;
  for (; i < n; i += stride) dst[i] = f2bf(src[i]);
}

// ---------------- bf16 WMMA GEMM with TDM-staged, double-buffered tiles -----
// C[M,N] = A[M,K] * Bw[N,K]^T (+bias). 256 thr = 8 waves (2x4); block tile
// 128x256; wave tile 64x64 -> 16 WMMA per K-step vs 16 ds_load_b128.
template <bool BIAS, bool OUT_BF16>
__global__ __launch_bounds__(256) void gemm_bf16(
    const unsigned short* __restrict__ A, const unsigned short* __restrict__ Bw,
    const float* __restrict__ bias, void* __restrict__ Cout, int M, int N,
    int K) {
  __shared__ unsigned short As[2][128 * 32];  // 8KB each
  __shared__ unsigned short Bs[2][256 * 32];  // 16KB each

  int wave = threadIdx.x >> 5;
  int lane = threadIdx.x & 31;
  int wm = wave & 1, wn = wave >> 1;
  int bm = blockIdx.x * 128, bn = blockIdx.y * 256;

  v8f acc[4][4];
#pragma unroll
  for (int i = 0; i < 4; ++i)
#pragma unroll
    for (int j = 0; j < 4; ++j) acc[i][j] = zero8();

  int nk = K / 32;
  if (wave == 0) {  // prime tile 0
    tdm_load_2d(lds_off(&As[0][0]), A + (size_t)bm * K, 32, 128, K);
    tdm_load_2d(lds_off(&Bs[0][0]), Bw + (size_t)bn * K, 32, 256, K);
  }

  for (int t = 0; t < nk; ++t) {
    int cur = t & 1;
    if (wave == 0) {
      if (t + 1 < nk) {  // issue next tile into the other buffer
        int k0 = (t + 1) * 32;
        tdm_load_2d(lds_off(&As[cur ^ 1][0]), A + (size_t)bm * K + k0, 32, 128, K);
        tdm_load_2d(lds_off(&Bs[cur ^ 1][0]), Bw + (size_t)bn * K + k0, 32, 256, K);
        __builtin_amdgcn_s_wait_tensorcnt(2);  // tile t done, t+1 in flight
      } else {
        __builtin_amdgcn_s_wait_tensorcnt(0);
      }
    }
    __syncthreads();

    v16bf a[4], b[4];
#pragma unroll
    for (int i = 0; i < 4; ++i)
      a[i] = load_frag_rm(&As[cur][0], 32, wm * 64 + 16 * i, 0);
#pragma unroll
    for (int j = 0; j < 4; ++j)
      b[j] = load_frag_rm(&Bs[cur][0], 32, wn * 64 + 16 * j, 0);
#pragma unroll
    for (int i = 0; i < 4; ++i)
#pragma unroll
      for (int j = 0; j < 4; ++j) acc[i][j] = wmma_bf16(a[i], b[j], acc[i][j]);

    __syncthreads();  // reads done before buffer is overwritten next iter
  }

  int c = lane & 15, half = lane >> 4;
#pragma unroll
  for (int i = 0; i < 4; ++i)
#pragma unroll
    for (int j = 0; j < 4; ++j)
#pragma unroll
      for (int r = 0; r < 8; ++r) {
        int row = bm + wm * 64 + 16 * i + r + 8 * half;
        int col = bn + wn * 64 + 16 * j + c;
        float v = acc[i][j][r];
        if (BIAS) v += bias[col];
        if (OUT_BF16)
          ((unsigned short*)Cout)[(size_t)row * N + col] = f2bf(v);
        else
          ((float*)Cout)[(size_t)row * N + col] = v;
      }
}

// ---------------- flash attention ----------------
// block = one (b,h) and 256 q rows; wave = 32 q rows (two 16-row tiles).
// K/V 32x64 tiles TDM-staged to LDS once per block, double-buffered; each
// staged K/V fragment now feeds 2 WMMAs (16 WMMA per 32-KV step per wave).
__global__ __launch_bounds__(256) void attn_bf16(
    const unsigned short* __restrict__ qkv,  // [B*S, 3D] bf16
    unsigned short* __restrict__ obf) {      // [B*S, D]  bf16
  __shared__ unsigned short Ks[2][32 * 64];   // 4KB each
  __shared__ unsigned short Vs[2][32 * 64];
  __shared__ unsigned short Pt[8][2 * 16 * 32];  // per-wave P tiles (2x1KB)

  int wave = threadIdx.x >> 5;
  int lane = threadIdx.x & 31;
  const int CH = SS / 256;  // q-chunks per (b,h)
  int ch = blockIdx.x % CH;
  int h = (blockIdx.x / CH) % HH;
  int b = blockIdx.x / (CH * HH);
  int q0 = ch * 256 + wave * 32;

  const unsigned short* base = qkv + (size_t)b * SS * E3;
  const unsigned short* Qp = base + h * DHH;
  const unsigned short* Kp = base + DD + h * DHH;
  const unsigned short* Vp = base + 2 * DD + h * DHH;
  const int ld = E3;

  v16bf qa[2][2];
#pragma unroll
  for (int rt = 0; rt < 2; ++rt) {
    qa[rt][0] = load_frag_rm(Qp, ld, q0 + 16 * rt, 0);
    qa[rt][1] = load_frag_rm(Qp, ld, q0 + 16 * rt, 32);
  }

  v8f o[2][4];
  float m[2][8], l[2][8];
#pragma unroll
  for (int rt = 0; rt < 2; ++rt) {
#pragma unroll
    for (int f = 0; f < 4; ++f) o[rt][f] = zero8();
#pragma unroll
    for (int r = 0; r < 8; ++r) { m[rt][r] = -3.0e38f; l[rt][r] = 0.0f; }
  }

  const float scale = 0.125f;  // 1/sqrt(64)
  const int nt = SS / 32;

  if (wave == 0) {  // prime tile 0
    tdm_load_2d(lds_off(&Ks[0][0]), Kp, 64, 32, E3);
    tdm_load_2d(lds_off(&Vs[0][0]), Vp, 64, 32, E3);
  }

  for (int t = 0; t < nt; ++t) {
    int cur = t & 1;
    int j = t * 32;
    if (wave == 0) {
      if (t + 1 < nt) {
        tdm_load_2d(lds_off(&Ks[cur ^ 1][0]), Kp + (size_t)(j + 32) * ld, 64, 32, E3);
        tdm_load_2d(lds_off(&Vs[cur ^ 1][0]), Vp + (size_t)(j + 32) * ld, 64, 32, E3);
        __builtin_amdgcn_s_wait_tensorcnt(2);
      } else {
        __builtin_amdgcn_s_wait_tensorcnt(0);
      }
    }
    __syncthreads();

    // K fragments shared across both q row-tiles
    v16bf kb00 = load_frag_rm(&Ks[cur][0], 64, 0, 0);
    v16bf kb01 = load_frag_rm(&Ks[cur][0], 64, 0, 32);
    v16bf kb10 = load_frag_rm(&Ks[cur][0], 64, 16, 0);
    v16bf kb11 = load_frag_rm(&Ks[cur][0], 64, 16, 32);

#pragma unroll
    for (int rt = 0; rt < 2; ++rt) {
      v8f s0 = zero8(), s1 = zero8();
      s0 = wmma_bf16(qa[rt][0], kb00, s0);
      s0 = wmma_bf16(qa[rt][1], kb01, s0);
      s1 = wmma_bf16(qa[rt][0], kb10, s1);
      s1 = wmma_bf16(qa[rt][1], kb11, s1);

      // online softmax; row r lives in vgpr r of each C frag (lane-half = +8)
#pragma unroll
      for (int r = 0; r < 8; ++r) {
        float x0 = s0[r] * scale, x1 = s1[r] * scale;
        float mx = row_max16(fmaxf(x0, x1));
        float mnew = fmaxf(m[rt][r], mx);
        float corr = __expf(m[rt][r] - mnew);
        float p0 = __expf(x0 - mnew);
        float p1 = __expf(x1 - mnew);
        float rs = row_sum16(p0 + p1);
        l[rt][r] = l[rt][r] * corr + rs;
        m[rt][r] = mnew;
#pragma unroll
        for (int f = 0; f < 4; ++f) o[rt][f][r] *= corr;
        int row = r + 8 * (lane >> 4);
        int col = lane & 15;
        Pt[wave][rt * 512 + row * 32 + col] = f2bf(p0);
        Pt[wave][rt * 512 + row * 32 + col + 16] = f2bf(p1);
      }
    }
    asm volatile("s_wait_dscnt 0" ::: "memory");  // intra-wave LDS RAW

    v16bf pa0 = load_frag_rm(&Pt[wave][0], 32, 0, 0);
    v16bf pa1 = load_frag_rm(&Pt[wave][512], 32, 0, 0);
#pragma unroll
    for (int f = 0; f < 4; ++f) {  // V fragment shared across both row tiles
      v16bf vb = load_bfrag_kn(&Vs[cur][0], 64, 0, 16 * f);
      o[0][f] = wmma_bf16(pa0, vb, o[0][f]);
      o[1][f] = wmma_bf16(pa1, vb, o[1][f]);
    }
    __syncthreads();  // all reads of K/V buffer done before TDM overwrites it
  }

  int c = lane & 15, half = lane >> 4;
#pragma unroll
  for (int rt = 0; rt < 2; ++rt)
#pragma unroll
    for (int f = 0; f < 4; ++f)
#pragma unroll
      for (int r = 0; r < 8; ++r) {
        int row = q0 + 16 * rt + r + 8 * half;
        int col = h * DHH + 16 * f + c;
        float v = o[rt][f][r] * (1.0f / l[rt][r]);
        obf[(size_t)(b * SS + row) * DD + col] = f2bf(v);
      }
}

// ---------------- launch ----------------
extern "C" void kernel_launch(void* const* d_in, const int* in_sizes, int n_in,
                              void* d_out, int out_size, void* d_ws,
                              size_t ws_size, hipStream_t stream) {
  const float* x = (const float*)d_in[0];      // [4,2048,1024]
  const float* w_in = (const float*)d_in[1];   // [3072,1024]
  const float* w_out = (const float*)d_in[2];  // [1024,1024]
  const float* b_out = (const float*)d_in[3];  // [1024]
  float* out = (float*)d_out;                  // [4,2048,1024]

  const int M = BB * SS;      // 8192
  const int nx = M * DD;      // 8388608
  const int nwi = E3 * DD;    // 3145728
  const int nwo = DD * DD;    // 1048576
  const int nqkv = M * E3;    // 25165824

  unsigned short* ws = (unsigned short*)d_ws;
  unsigned short* x_bf = ws;   ws += nx;
  unsigned short* wi_bf = ws;  ws += nwi;
  unsigned short* wo_bf = ws;  ws += nwo;
  unsigned short* qkv_bf = ws; ws += nqkv;
  unsigned short* attn_bf = ws;

  cvt_f32_bf16<<<2048, 256, 0, stream>>>(x, x_bf, nx);
  cvt_f32_bf16<<<2048, 256, 0, stream>>>(w_in, wi_bf, nwi);
  cvt_f32_bf16<<<1024, 256, 0, stream>>>(w_out, wo_bf, nwo);

  {  // qkv = x @ w_in^T  (8192 x 3072 x 1024), bf16 out
    dim3 grid(M / 128, E3 / 256);
    gemm_bf16<false, true><<<grid, 256, 0, stream>>>(x_bf, wi_bf, nullptr,
                                                     (void*)qkv_bf, M, E3, DD);
  }

  // flash attention -> attn_bf [8192, 1024]; 1 block per (b,h,256-row chunk)
  attn_bf16<<<BB * HH * (SS / 256), 256, 0, stream>>>(qkv_bf, attn_bf);

  {  // out = attn @ w_out^T + b_out  (8192 x 1024 x 1024), fp32 out
    dim3 grid(M / 128, DD / 256);
    gemm_bf16<true, false><<<grid, 256, 0, stream>>>(attn_bf, wo_bf, b_out,
                                                     (void*)out, M, DD, DD);
  }
}
